// LocalWindowAttention_33105607918093
// MI455X (gfx1250) — matrix operand order repaired and checked by reading, not compiled, not run on hardware
//
#include <hip/hip_runtime.h>
#include <hip/hip_bf16.h>

#define DIM 2048
#define H   16
#define HD  128
#define W   16
#define NT  16384   // B*S = 4*4096

typedef __attribute__((ext_vector_type(16))) __bf16 v16bf;
typedef __attribute__((ext_vector_type(8)))  __bf16 v8bf;
typedef __attribute__((ext_vector_type(8)))  float  v8f;

union ABu { v16bf v; v8bf h[2]; };

__device__ __forceinline__ unsigned short f2bf_rne(float f) {
  unsigned int u = __float_as_uint(f);
  u += 0x7FFFu + ((u >> 16) & 1u);
  return (unsigned short)(u >> 16);
}

// ---------------- Kernel A: f32 -> bf16 convert (vectorized x4) ----------------
__global__ __launch_bounds__(256)
void cvt_f32_to_bf16(const float* __restrict__ src, unsigned short* __restrict__ dst, int n4) {
  int i = blockIdx.x * blockDim.x + threadIdx.x;
  if (i >= n4) return;
  float4 f = reinterpret_cast<const float4*>(src)[i];
  ushort4 o;
  o.x = f2bf_rne(f.x); o.y = f2bf_rne(f.y);
  o.z = f2bf_rne(f.z); o.w = f2bf_rne(f.w);
  reinterpret_cast<ushort4*>(dst)[i] = o;
}

// ---------------- Kernel B: per-window QKV + RoPE + attention ----------------
// grid = 256 window-slabs * 16 heads; block = 128 threads = 4 waves.
// The 4 waves are 4 DIFFERENT WINDOWS of the SAME head, so the per-k-step
// weight tile (128 x 32 bf16) is staged once in LDS and shared by all 4 waves
// (4x reduction of L2 weight traffic: 24 GB -> 6 GB).
__global__ __launch_bounds__(128)
void attn_window_kernel(const unsigned short* __restrict__ xbf,
                        const unsigned short* __restrict__ wqbf,
                        const unsigned short* __restrict__ wkbf,
                        const unsigned short* __restrict__ wvbf,
                        const float* __restrict__ rope,
                        unsigned short* __restrict__ outbf) {
  // 4 * 13KB per-wave staging + 128x40 bf16 shared B tile (row-padded 32->40
  // so WMMA B-operand ds_load_b128 is bank-conflict-free) = 62 KB
  __shared__ __align__(16) unsigned short smem[4 * 6656 + 128 * 40];
  const int tid  = threadIdx.x;
  const int wid  = tid >> 5;
  const int lane = tid & 31;
  const int lo   = lane & 15;
  const int hi   = lane >> 4;
  const int slab = blockIdx.x >> 4;          // 256 slabs of 4 windows
  const int h    = blockIdx.x & 15;          // head (same for all 4 waves)
  const int p    = slab * 4 + wid;           // this wave's window

  unsigned short* qlds = smem + wid * 6656;  // 16x128 bf16
  unsigned short* klds = qlds + 2048;        // 16x128 bf16
  unsigned short* vlds = qlds + 4096;        // 128x16 bf16 (v transposed)
  unsigned short* alds = qlds + 6144;        // 16x32 bf16 (attn, K-padded)
  unsigned short* bbuf = smem + 4 * 6656;    // 128 x 40 shared weight tile

  // RoPE cos/sin per lane: row w = r+8*hi, dim d = n*16+lo (d < 64)
  float cc[4][8], ss[4][8];
  #pragma unroll
  for (int n = 0; n < 4; ++n)
    #pragma unroll
    for (int r = 0; r < 8; ++r) {
      float f = rope[(r + 8 * hi) * (HD / 2) + n * 16 + lo];
      cc[n][r] = __cosf(f);
      ss[n][r] = __sinf(f);
    }

  const size_t xrow = (size_t)(p * W + lo) * DIM;

  // ---- Q, K, V projections: 8 tiles of 16x16 per wave, K-loop over 2048 ----
  for (int m = 0; m < 3; ++m) {
    const unsigned short* wm = (m == 0) ? wqbf : (m == 1) ? wkbf : wvbf;
    v8f acc[8] = {};
    for (int k0 = 0; k0 < DIM; k0 += 32) {
      // cooperative stage of B tile: thread t -> weight row h*128+t, 64 bytes
      {
        const uint4* g = reinterpret_cast<const uint4*>(
            wm + (size_t)(h * HD + tid) * DIM + k0);
        uint4* s = reinterpret_cast<uint4*>(bbuf + tid * 40);
        s[0] = g[0]; s[1] = g[1]; s[2] = g[2]; s[3] = g[3];
      }
      __syncthreads();
      ABu a;
      const v8bf* pa = reinterpret_cast<const v8bf*>(xbf + xrow + k0 + 8 * hi);
      a.h[0] = pa[0];          // K = k0+8*hi .. +7
      a.h[1] = pa[2];          // K = k0+16+8*hi .. +7
      #pragma unroll
      for (int n = 0; n < 8; ++n) {
        ABu b;
        const v8bf* pb = reinterpret_cast<const v8bf*>(
            bbuf + (n * 16 + lo) * 40 + 8 * hi);
        b.h[0] = pb[0]; b.h[1] = pb[2];  // K = k0+8*hi..+7, k0+16+8*hi..+7
        acc[n] = __builtin_amdgcn_wmma_f32_16x16x32_bf16(
            false, a.v, false, b.v, (short)0, acc[n], false, false);
      }
      __syncthreads();
    }
    if (m < 2) {
      // in-register RoPE on accumulators (pairs d, d+64 -> tiles n, n+4)
      #pragma unroll
      for (int n = 0; n < 4; ++n)
        #pragma unroll
        for (int r = 0; r < 8; ++r) {
          float t1 = acc[n][r], t2 = acc[n + 4][r];
          acc[n][r]     = t1 * cc[n][r] - t2 * ss[n][r];
          acc[n + 4][r] = t2 * cc[n][r] + t1 * ss[n][r];
        }
      unsigned short* dst = (m == 0) ? qlds : klds;
      #pragma unroll
      for (int n = 0; n < 8; ++n)
        #pragma unroll
        for (int r = 0; r < 8; ++r)
          dst[(r + 8 * hi) * HD + n * 16 + lo] = f2bf_rne(acc[n][r]);
    } else {
      // V staged transposed: vT[d][t]
      #pragma unroll
      for (int n = 0; n < 8; ++n)
        #pragma unroll
        for (int r = 0; r < 8; ++r)
          vlds[(n * 16 + lo) * W + r + 8 * hi] = f2bf_rne(acc[n][r]);
    }
  }

  // ---- scores = q @ k^T * 1/sqrt(HD) ----
  v8f s = {};
  for (int k0 = 0; k0 < HD; k0 += 32) {
    ABu a, b;
    const v8bf* pa = reinterpret_cast<const v8bf*>(qlds + lo * HD + k0 + 8 * hi);
    a.h[0] = pa[0]; a.h[1] = pa[2];
    const v8bf* pb = reinterpret_cast<const v8bf*>(klds + lo * HD + k0 + 8 * hi);
    b.h[0] = pb[0]; b.h[1] = pb[2];
    s = __builtin_amdgcn_wmma_f32_16x16x32_bf16(
        false, a.v, false, b.v, (short)0, s, false, false);
  }

  // ---- softmax per row (row spread over 16 lanes of a half-wave) ----
  const float scale = 0.08838834764831845f;  // 1/sqrt(128)
  #pragma unroll
  for (int r = 0; r < 8; ++r) {
    float v = s[r] * scale;
    float mx = v;
    #pragma unroll
    for (int off = 1; off < 16; off <<= 1)
      mx = fmaxf(mx, __shfl_xor(mx, off, 32));
    float e = __expf(v - mx);
    float sum = e;
    #pragma unroll
    for (int off = 1; off < 16; off <<= 1)
      sum += __shfl_xor(sum, off, 32);
    float a = e / sum;
    alds[(r + 8 * hi) * 32 + lo]      = f2bf_rne(a);
    alds[(r + 8 * hi) * 32 + 16 + lo] = 0;       // explicit K-pad zeros (no 0*NaN)
  }

  // ---- out = attn @ v (K=16 zero-padded to 32) ----
  v8f o[8] = {};
  {
    ABu a;
    const v8bf* pa = reinterpret_cast<const v8bf*>(alds + lo * 32 + 8 * hi);
    a.h[0] = pa[0];          // K = 8*hi..+7 (real attn)
    a.h[1] = pa[2];          // K = 16+8*hi..+7 (zeros)
    #pragma unroll
    for (int n = 0; n < 8; ++n) {
      ABu b;
      const v8bf* pb = reinterpret_cast<const v8bf*>(vlds + (n * 16 + lo) * W + 8 * hi);
      b.h[0] = pb[0]; b.h[1] = pb[2];  // K = 8*hi..+7; second half hits A's zeros
      o[n] = __builtin_amdgcn_wmma_f32_16x16x32_bf16(
          false, a.v, false, b.v, (short)0, o[n], false, false);
    }
  }

  // ---- stage output (reuse q region) then coalesced b128 store ----
  #pragma unroll
  for (int n = 0; n < 8; ++n)
    #pragma unroll
    for (int r = 0; r < 8; ++r)
      qlds[(r + 8 * hi) * HD + n * 16 + lo] = f2bf_rne(o[n][r]);
  #pragma unroll
  for (int i = 0; i < 8; ++i) {
    int idx = i * 32 + lane;          // 256 chunks of 8 bf16
    int rr  = idx >> 4;
    int c8  = (idx & 15) * 8;
    uint4 d = *reinterpret_cast<const uint4*>(qlds + rr * HD + c8);
    *reinterpret_cast<uint4*>(outbf + (size_t)(p * W + rr) * DIM + h * HD + c8) = d;
  }
}

// ---------------- Kernel C: output projection y = out @ wo^T ----------------
// grid (128, 16): 128x128 tile per 8-wave WG; wave = 16 rows x 128 cols.
// All 8 waves read the same wo rows -> WGP$ absorbs B re-reads.
__global__ __launch_bounds__(256)
void proj_kernel(const unsigned short* __restrict__ outbf,
                 const unsigned short* __restrict__ wobf,
                 float* __restrict__ y) {
  const int tid  = threadIdx.x;
  const int wid  = tid >> 5;
  const int lane = tid & 31;
  const int lo   = lane & 15;
  const int hi   = lane >> 4;
  const int m0   = blockIdx.x * 128 + wid * 16;
  const int n0   = blockIdx.y * 128;

  v8f acc[8] = {};
  for (int k0 = 0; k0 < DIM; k0 += 32) {
    ABu a;
    const v8bf* pa = reinterpret_cast<const v8bf*>(
        outbf + (size_t)(m0 + lo) * DIM + k0 + 8 * hi);
    a.h[0] = pa[0]; a.h[1] = pa[2];
    #pragma unroll
    for (int n = 0; n < 8; ++n) {
      ABu b;
      const v8bf* pb = reinterpret_cast<const v8bf*>(
          wobf + (size_t)(n0 + n * 16 + lo) * DIM + k0 + 8 * hi);
      b.h[0] = pb[0]; b.h[1] = pb[2];
      acc[n] = __builtin_amdgcn_wmma_f32_16x16x32_bf16(
          false, a.v, false, b.v, (short)0, acc[n], false, false);
    }
  }
  #pragma unroll
  for (int n = 0; n < 8; ++n)
    #pragma unroll
    for (int r = 0; r < 8; ++r)
      y[(size_t)(m0 + r + 8 * hi) * DIM + n0 + n * 16 + lo] = acc[n][r];
}

// ---------------- host launcher ----------------
extern "C" void kernel_launch(void* const* d_in, const int* in_sizes, int n_in,
                              void* d_out, int out_size, void* d_ws, size_t ws_size,
                              hipStream_t stream) {
  (void)in_sizes; (void)n_in; (void)out_size; (void)ws_size;
  const float* x    = (const float*)d_in[0];
  const float* rope = (const float*)d_in[1];
  const float* wq   = (const float*)d_in[2];
  const float* wk   = (const float*)d_in[3];
  const float* wv   = (const float*)d_in[4];
  const float* wo   = (const float*)d_in[5];
  float* y = (float*)d_out;

  char* ws = (char*)d_ws;
  const size_t XB = (size_t)NT * DIM * 2;       // 67,108,864
  const size_t WB = (size_t)DIM * DIM * 2;      //  8,388,608
  unsigned short* xbf  = (unsigned short*)(ws);
  unsigned short* wqbf = (unsigned short*)(ws + XB);
  unsigned short* wkbf = (unsigned short*)(ws + XB + WB);
  unsigned short* wvbf = (unsigned short*)(ws + XB + 2 * WB);
  unsigned short* wobf = (unsigned short*)(ws + XB + 3 * WB);
  unsigned short* obf  = (unsigned short*)(ws + XB + 4 * WB);

  int n4 = NT * DIM / 4;
  cvt_f32_to_bf16<<<(n4 + 255) / 256, 256, 0, stream>>>(x, xbf, n4);
  int w4 = DIM * DIM / 4;
  cvt_f32_to_bf16<<<(w4 + 255) / 256, 256, 0, stream>>>(wq, wqbf, w4);
  cvt_f32_to_bf16<<<(w4 + 255) / 256, 256, 0, stream>>>(wk, wkbf, w4);
  cvt_f32_to_bf16<<<(w4 + 255) / 256, 256, 0, stream>>>(wv, wvbf, w4);
  cvt_f32_to_bf16<<<(w4 + 255) / 256, 256, 0, stream>>>(wo, wobf, w4);

  attn_window_kernel<<<4096, 128, 0, stream>>>(xbf, wqbf, wkbf, wvbf, rope, obf);

  dim3 gc(128, 16);
  proj_kernel<<<gc, 256, 0, stream>>>(obf, wobf, y);
}
